// Renderer_62835371540525
// MI455X (gfx1250) — compile-verified, matrix operand order checked by reading, not run on hardware
//
#include <hip/hip_runtime.h>
#include <math.h>

typedef __attribute__((ext_vector_type(16))) _Float16 v16h;
typedef __attribute__((ext_vector_type(8)))  float    v8f;

#define HID       128
#define RAY_STEPS 128
#define N_SECANT  8
#define STEPS_IN  64
#define STEPS_OUT 32
#define FULL      96
#define TAU_C     0.5f
#define EPS_C     1e-6f
// DELTA = max(2*exp(-1.5e-5*100000), 0.1) = 2*exp(-1.5)
#define DELTA_C   0.44626032029685964f

// Branchless fast softplus: max(x,0) + log(1 + exp(-|x|))  (TRANS pipeline)
__device__ __forceinline__ float softplusf(float x) {
    return fmaxf(x, 0.f) + __logf(1.f + __expf(-fabsf(x)));
}
// Fast sigmoid: rcp(1 + exp(-x))
__device__ __forceinline__ float sigmoidf(float x) {
    return __frcp_rn(1.f + __expf(-x));
}

// ---------------------------------------------------------------------------
// Kernel 1: M = inv(C @ W @ S) and cam0 = M[:,3]   (single thread)
// ---------------------------------------------------------------------------
__device__ void inv4x4(const float* A, float* out) {
    float a[4][8];
    for (int i = 0; i < 4; ++i)
        for (int j = 0; j < 4; ++j) { a[i][j] = A[i*4+j]; a[i][4+j] = (i == j) ? 1.f : 0.f; }
    for (int c = 0; c < 4; ++c) {
        int p = c;
        for (int r = c + 1; r < 4; ++r) if (fabsf(a[r][c]) > fabsf(a[p][c])) p = r;
        if (p != c) for (int j = 0; j < 8; ++j) { float t = a[c][j]; a[c][j] = a[p][j]; a[p][j] = t; }
        float inv = 1.f / a[c][c];
        for (int j = 0; j < 8; ++j) a[c][j] *= inv;
        for (int r = 0; r < 4; ++r) if (r != c) {
            float f = a[r][c];
            for (int j = 0; j < 8; ++j) a[r][j] -= f * a[c][j];
        }
    }
    for (int i = 0; i < 4; ++i)
        for (int j = 0; j < 4; ++j) out[i*4+j] = a[i][4+j];
}

__global__ void k_init(const float* __restrict__ C, const float* __restrict__ W,
                       const float* __restrict__ S, float* __restrict__ ws) {
    if (threadIdx.x != 0 || blockIdx.x != 0) return;
    float T1[16], T[16], M[16];
    for (int i = 0; i < 4; ++i)
        for (int j = 0; j < 4; ++j) {
            float s = 0.f;
            for (int k = 0; k < 4; ++k) s += C[i*4+k] * W[k*4+j];
            T1[i*4+j] = s;
        }
    for (int i = 0; i < 4; ++i)
        for (int j = 0; j < 4; ++j) {
            float s = 0.f;
            for (int k = 0; k < 4; ++k) s += T1[i*4+k] * S[k*4+j];
            T[i*4+j] = s;
        }
    inv4x4(T, M);
    for (int i = 0; i < 16; ++i) ws[i] = M[i];
    ws[16] = M[3];  ws[17] = M[7];  ws[18] = M[11];   // cam0 = M @ (0,0,0,1)
}

// ---------------------------------------------------------------------------
// Kernel 2: per-ray direction + sphere intersection
// ---------------------------------------------------------------------------
__global__ void k_rays(const float* __restrict__ pixels, const float* __restrict__ ws,
                       float* __restrict__ raydir, float* __restrict__ dfar,
                       float* __restrict__ maskint, int N) {
    int i = blockIdx.x * blockDim.x + threadIdx.x;
    if (i >= N) return;
    float cx = ws[16], cy = ws[17], cz = ws[18];
    float px = pixels[2*i], py = pixels[2*i+1];
    // ray = M @ (px,py,1,1) - cam0  (the M[:,3] term cancels)
    float rx = ws[0]*px + ws[1]*py + ws[2];
    float ry = ws[4]*px + ws[5]*py + ws[6];
    float rz = ws[8]*px + ws[9]*py + ws[10];
    float inv = rsqrtf(rx*rx + ry*ry + rz*rz);
    rx *= inv; ry *= inv; rz *= inv;
    float rcd   = rx*cx + ry*cy + rz*cz;
    float c2    = cx*cx + cy*cy + cz*cz;
    float under = rcd*rcd - (c2 - 1.f);
    bool  m     = under > 0.f;
    float s     = sqrtf(m ? under : 1.f);
    float dF    = m ? fmaxf(s - rcd, 0.f) : 0.f;
    raydir[3*i+0] = rx; raydir[3*i+1] = ry; raydir[3*i+2] = rz;
    dfar[i]    = dF;
    maskint[i] = m ? 1.f : 0.f;
}

// ---------------------------------------------------------------------------
// WMMA fragment helpers (shapes per CDNA5 ISA 7.12.2, wave32)
// ---------------------------------------------------------------------------
// B 32x16 f16: lanes 0-15 hold K=0..7 (halves 0..7) of column n=lane&15;
// K>=8 lives in lanes>=16 / halves 8..15 and is always zero here (Krows<=6).
template <int KROWS>
__device__ __forceinline__ v16h build_B(const float* __restrict__ Wm, int colBase, int lane) {
    v16h b;
#pragma unroll
    for (int h = 0; h < 16; ++h) b[h] = (_Float16)0.f;
    if (lane < 16) {
        int col = colBase + (lane & 15);
#pragma unroll
        for (int k = 0; k < KROWS; ++k) b[k] = (_Float16)Wm[k * HID + col];
    }
    return b;
}

// Single-point occupancy (used by secant): hidden units spread over 32 lanes.
__device__ __forceinline__ float occ_point(float px, float py, float pz,
                                           const float* __restrict__ W1,
                                           const float* __restrict__ b1,
                                           const float* __restrict__ W2,
                                           float b2v, int lane) {
    float acc = 0.f;
#pragma unroll
    for (int q = 0; q < 4; ++q) {
        int h = lane * 4 + q;
        float s = px * W1[h] + py * W1[HID + h] + pz * W1[2*HID + h] + b1[h];
        acc += softplusf(s) * W2[h];
    }
#pragma unroll
    for (int m = 1; m < 32; m <<= 1) acc += __shfl_xor(acc, m, 32);
    return sigmoidf(acc + b2v) - TAU_C;
}

// ---------------------------------------------------------------------------
// Kernel 3: ray march (WMMA occupancy, 8 tiles of 16 samples) + secant
// ---------------------------------------------------------------------------
__global__ __launch_bounds__(128) void k_march(
    const float* __restrict__ W1, const float* __restrict__ b1,
    const float* __restrict__ W2, const float* __restrict__ b2,
    const float* __restrict__ ws, const float* __restrict__ raydir,
    const float* __restrict__ dfar, const float* __restrict__ maskint,
    float* __restrict__ d_i, int N) {
    int lane = threadIdx.x & 31;
    int ray  = blockIdx.x * (blockDim.x >> 5) + (threadIdx.x >> 5);
    if (ray >= N) return;
    int n = lane & 15;
    float cx = ws[16], cy = ws[17], cz = ws[18];
    float rx = raydir[3*ray+0], ry = raydir[3*ray+1], rz = raydir[3*ray+2];
    float dF = dfar[ray];
    bool mint = maskint[ray] > 0.5f;
    float b2v = b2[0];

    v16h BW[8];
    float cb[8], w2v[8];
#pragma unroll
    for (int j = 0; j < 8; ++j) {
        BW[j]  = build_B<3>(W1, j * 16, lane);
        cb[j]  = b1[j*16 + n];
        w2v[j] = W2[j*16 + n];
    }

    bool found = false;
    float prev = 0.f, val0 = 0.f, f_lo = 0.f, f_hi = 0.f;
    int ind = 0;

    for (int t = 0; t < 8; ++t) {
        // A fragment: row M = lane&15 is sample t*16+M
        v16h a;
#pragma unroll
        for (int h = 0; h < 16; ++h) a[h] = (_Float16)0.f;
        {
            int s = t * 16 + n;
            float d = dF * (float)s * (1.f / 127.f);
            if (lane < 16) {
                a[0] = (_Float16)(cx + rx * d);
                a[1] = (_Float16)(cy + ry * d);
                a[2] = (_Float16)(cz + rz * d);
            }
        }
        float acc[8];
#pragma unroll
        for (int r = 0; r < 8; ++r) acc[r] = 0.f;
#pragma unroll
        for (int j = 0; j < 8; ++j) {
            v8f c;
#pragma unroll
            for (int r = 0; r < 8; ++r) c[r] = cb[j];
            v8f dm = __builtin_amdgcn_wmma_f32_16x16x32_f16(false, a, false, BW[j],
                                                            (short)0, c, false, false);
#pragma unroll
            for (int r = 0; r < 8; ++r) acc[r] += softplusf(dm[r]) * w2v[j];
        }
#pragma unroll
        for (int r = 0; r < 8; ++r) {
            acc[r] += __shfl_xor(acc[r], 1, 32);
            acc[r] += __shfl_xor(acc[r], 2, 32);
            acc[r] += __shfl_xor(acc[r], 4, 32);
            acc[r] += __shfl_xor(acc[r], 8, 32);
            acc[r] = sigmoidf(acc[r] + b2v) - TAU_C;
        }
        // wave-uniform serial sign scan (first crossing)
#pragma unroll
        for (int m = 0; m < 16; ++m) {
            float v = __shfl(acc[m & 7], (m >> 3) * 16, 32);
            int i = t * 16 + m;
            if (i == 0) {
                val0 = v;
            } else if (!found && (prev * v) < 0.f) {
                found = true; ind = i - 1; f_lo = prev; f_hi = v;
            }
            prev = v;
        }
    }

    bool mask = found && (f_lo < 0.f) && mint && (val0 < 0.f);
    float out;
    if (mask) {
        float dl = dF * (float)ind * (1.f / 127.f);
        float dh = dF * (float)(ind + 1) * (1.f / 127.f);
        float fl = f_lo, fh = f_hi;
        float den = fh - fl;
        float dp = (fabsf(den) > 1e-12f) ? (-fl * (dh - dl) / den + dl) : 0.5f * (dl + dh);
        for (int it = 0; it < N_SECANT; ++it) {
            float fm = occ_point(cx + rx*dp, cy + ry*dp, cz + rz*dp, W1, b1, W2, b2v, lane);
            bool low = fm < 0.f;
            dl = low ? dp : dl;  fl = low ? fm : fl;
            dh = low ? dh : dp;  fh = low ? fh : fm;
            den = fh - fl;
            dp = (fabsf(den) > 1e-12f) ? (-fl * (dh - dl) / den + dl) : 0.5f * (dl + dh);
        }
        out = dp;
    } else {
        out = (val0 < 0.f) ? __builtin_inff() : 0.f;
    }
    if (lane == 0) d_i[ray] = out;
}

// ---------------------------------------------------------------------------
// Kernel 4: depth interval selection (merge == jnp.sort of concat)
// ---------------------------------------------------------------------------
__global__ void k_intervals(const float* __restrict__ d_i, const float* __restrict__ dfar,
                            float* __restrict__ d_all, int N) {
    int ray = blockIdx.x * blockDim.x + threadIdx.x;
    if (ray >= N) return;
    float d  = d_i[ray];
    float dF = dfar[ray];
    bool mzero = (d == 0.f);
    bool mpred = __builtin_isfinite(d);
    float dists = mpred ? d : 1.f;
    if (mzero) dists = 0.f;
    bool obj = mpred && !mzero;
    float* out = d_all + (size_t)ray * FULL;
    if (!obj) {
        for (int k = 0; k < FULL; ++k) out[k] = dF * (float)k * (1.f / 95.f);
    } else {
        float dnp = fmaxf(dists - DELTA_C, 0.f);
        float dfp = fminf(dists + DELTA_C, dF);
        bool asc = dfp >= dnp;
        int i = 0, j = 0;
        for (int k = 0; k < FULL; ++k) {
            float av = (i < STEPS_OUT) ? dnp * (float)i * (1.f / 31.f) : __builtin_inff();
            float bv;
            if (j < STEPS_IN) {
                int jj = asc ? j : (STEPS_IN - 1 - j);
                bv = dnp + (dfp - dnp) * (float)jj * (1.f / 63.f);
            } else {
                bv = __builtin_inff();
            }
            if (av <= bv) { out[k] = av; ++i; } else { out[k] = bv; ++j; }
        }
    }
}

// ---------------------------------------------------------------------------
// Kernel 5: shading (occ + color MLP via WMMA) + alpha compositing
// ---------------------------------------------------------------------------
__global__ __launch_bounds__(128) void k_shade(
    const float* __restrict__ W1,  const float* __restrict__ b1,
    const float* __restrict__ W2,  const float* __restrict__ b2,
    const float* __restrict__ Wc1, const float* __restrict__ bc1,
    const float* __restrict__ Wc2, const float* __restrict__ bc2,
    const float* __restrict__ ws,  const float* __restrict__ raydir,
    const float* __restrict__ d_all, float* __restrict__ out_rgb, int N) {
    int lane = threadIdx.x & 31;
    int ray  = blockIdx.x * (blockDim.x >> 5) + (threadIdx.x >> 5);
    if (ray >= N) return;
    int n = lane & 15;
    float cx = ws[16], cy = ws[17], cz = ws[18];
    float rx = raydir[3*ray+0], ry = raydir[3*ray+1], rz = raydir[3*ray+2];
    float b2v = b2[0];
    float bc2v0 = bc2[0], bc2v1 = bc2[1], bc2v2 = bc2[2];

    v16h BW[8], BC[8];
    float cb1v[8], cbc1v[8], w2v[8], wc2v0[8], wc2v1[8], wc2v2[8];
#pragma unroll
    for (int j = 0; j < 8; ++j) {
        BW[j] = build_B<3>(W1,  j * 16, lane);
        BC[j] = build_B<6>(Wc1, j * 16, lane);
        int col  = j * 16 + n;
        cb1v[j]  = b1[col];
        cbc1v[j] = bc1[col];
        w2v[j]   = W2[col];
        wc2v0[j] = Wc2[col * 3 + 0];
        wc2v1[j] = Wc2[col * 3 + 1];
        wc2v2[j] = Wc2[col * 3 + 2];
    }

    float trans = 1.f;
    float sum0 = 0.f, sum1 = 0.f, sum2 = 0.f;

    for (int t = 0; t < 6; ++t) {
        v16h a;
#pragma unroll
        for (int h = 0; h < 16; ++h) a[h] = (_Float16)0.f;
        {
            int k = t * 16 + n;
            float d = d_all[(size_t)ray * FULL + k];
            if (lane < 16) {
                a[0] = (_Float16)(cx + rx * d);
                a[1] = (_Float16)(cy + ry * d);
                a[2] = (_Float16)(cz + rz * d);
                a[3] = (_Float16)(-rx);
                a[4] = (_Float16)(-ry);
                a[5] = (_Float16)(-rz);
            }
        }
        float accA[8], accR0[8], accR1[8], accR2[8];
#pragma unroll
        for (int r = 0; r < 8; ++r) { accA[r] = 0.f; accR0[r] = 0.f; accR1[r] = 0.f; accR2[r] = 0.f; }
#pragma unroll
        for (int j = 0; j < 8; ++j) {
            v8f c1, c2;
#pragma unroll
            for (int r = 0; r < 8; ++r) { c1[r] = cb1v[j]; c2[r] = cbc1v[j]; }
            v8f h1 = __builtin_amdgcn_wmma_f32_16x16x32_f16(false, a, false, BW[j],
                                                            (short)0, c1, false, false);
            v8f f1 = __builtin_amdgcn_wmma_f32_16x16x32_f16(false, a, false, BC[j],
                                                            (short)0, c2, false, false);
#pragma unroll
            for (int r = 0; r < 8; ++r) {
                accA[r] += softplusf(h1[r]) * w2v[j];
                float f = fmaxf(f1[r], 0.f);
                accR0[r] += f * wc2v0[j];
                accR1[r] += f * wc2v1[j];
                accR2[r] += f * wc2v2[j];
            }
        }
#pragma unroll
        for (int r = 0; r < 8; ++r) {
#pragma unroll
            for (int off = 1; off < 16; off <<= 1) {
                accA[r]  += __shfl_xor(accA[r],  off, 32);
                accR0[r] += __shfl_xor(accR0[r], off, 32);
                accR1[r] += __shfl_xor(accR1[r], off, 32);
                accR2[r] += __shfl_xor(accR2[r], off, 32);
            }
            accA[r]  = sigmoidf(accA[r]  + b2v);
            accR0[r] = sigmoidf(accR0[r] + bc2v0);
            accR1[r] = sigmoidf(accR1[r] + bc2v1);
            accR2[r] = sigmoidf(accR2[r] + bc2v2);
        }
        // wave-uniform serial compositing over the 16 points of this tile
#pragma unroll
        for (int m = 0; m < 16; ++m) {
            int src = (m >> 3) * 16;
            float av = __shfl(accA[m & 7],  src, 32);
            float r0 = __shfl(accR0[m & 7], src, 32);
            float g0 = __shfl(accR1[m & 7], src, 32);
            float bb = __shfl(accR2[m & 7], src, 32);
            float w = av * trans;
            sum0 += w * r0; sum1 += w * g0; sum2 += w * bb;
            trans *= (1.f - av + EPS_C);
        }
    }
    if (lane == 0) {
        out_rgb[3*ray+0] = sum0;
        out_rgb[3*ray+1] = sum1;
        out_rgb[3*ray+2] = sum2;
    }
}

// ---------------------------------------------------------------------------
extern "C" void kernel_launch(void* const* d_in, const int* in_sizes, int n_in,
                              void* d_out, int out_size, void* d_ws, size_t ws_size,
                              hipStream_t stream) {
    const float* pixels = (const float*)d_in[0];
    const float* cam    = (const float*)d_in[1];
    const float* wm     = (const float*)d_in[2];
    const float* sm     = (const float*)d_in[3];
    const float* W1     = (const float*)d_in[4];
    const float* b1     = (const float*)d_in[5];
    const float* W2     = (const float*)d_in[6];
    const float* b2     = (const float*)d_in[7];
    const float* Wc1    = (const float*)d_in[8];
    const float* bc1    = (const float*)d_in[9];
    const float* Wc2    = (const float*)d_in[10];
    const float* bc2    = (const float*)d_in[11];

    int N = in_sizes[0] / 2;   // (B=1, N, 2)

    float* ws      = (float*)d_ws;
    float* raydir  = ws + 32;                  // 3*N
    float* dfar    = raydir + 3 * (size_t)N;   // N
    float* maskint = dfar + N;                 // N
    float* di      = maskint + N;              // N
    float* dall    = di + N;                   // N*96

    k_init<<<1, 32, 0, stream>>>(cam, wm, sm, ws);
    k_rays<<<(N + 255) / 256, 256, 0, stream>>>(pixels, ws, raydir, dfar, maskint, N);
    k_march<<<(N + 3) / 4, 128, 0, stream>>>(W1, b1, W2, b2, ws, raydir, dfar, maskint, di, N);
    k_intervals<<<(N + 255) / 256, 256, 0, stream>>>(di, dfar, dall, N);
    k_shade<<<(N + 3) / 4, 128, 0, stream>>>(W1, b1, W2, b2, Wc1, bc1, Wc2, bc2,
                                             ws, raydir, dall, (float*)d_out, N);
}
